// TwoDimensionalSSM_71975061946624
// MI455X (gfx1250) — matrix operand order repaired and checked by reading, not compiled, not run on hardware
//
#include <hip/hip_runtime.h>

typedef __attribute__((ext_vector_type(8)))  float  v8f;
typedef __attribute__((ext_vector_type(16))) __bf16 v16bf;
typedef __attribute__((ext_vector_type(8)))  __bf16 v8bf;

// S=1024 (Ls=32), B=16, E=1024, H=256 (4 dirs x 64), N=2, classes=64, rep=16
#define LS 32
#define NB 16
#define NE 1024

__device__ __forceinline__ float sigmoidf_(float v) {
  return 1.0f / (1.0f + __expf(-v));
}

__device__ __forceinline__ v16bf combine16(v8bf lo, v8bf hi) {
  union { struct { v8bf lo, hi; } s; v16bf v; } u;
  u.s.lo = lo; u.s.hi = hi;
  return u.v;
}

// ---------------------------------------------------------------- kernel 1
// 2D SSM impulse response: K[h][i][l], boundary corrections folded in.
__global__ __launch_bounds__(256) void compute_k_kernel(
    const float* __restrict__ A1g, const float* __restrict__ A2g,
    const float* __restrict__ A3g, const float* __restrict__ A4g,
    const float* __restrict__ B1g, const float* __restrict__ B2g,
    const float* __restrict__ C1g, const float* __restrict__ C2g,
    float* __restrict__ Kout) {
  const int h = threadIdx.x;
  const float scale = 0.70710678118654752f;  // sqrt(1/2)
  for (int n = 0; n < 2; ++n) {
    const int p = h * 2 + n;
    const float a1 = sigmoidf_(A1g[p]), a2 = sigmoidf_(A2g[p]);
    const float a3 = sigmoidf_(A3g[p]), a4 = sigmoidf_(A4g[p]);
    const float b1 = sigmoidf_(B1g[p]), b2 = sigmoidf_(B2g[p]);
    const float c1 = C1g[p], c2 = C2g[p];
    float xh[LS], xv[LS];
#pragma unroll
    for (int l = 0; l < LS; ++l) { xh[l] = 0.f; xv[l] = 0.f; }
    for (int i = 0; i < LS; ++i) {
      const float first = (i == 0) ? 1.0f : 0.0f;
#pragma unroll
      for (int l = 0; l < LS; ++l) xv[l] = a3 * xh[l] + a4 * xv[l];
      xv[0] += first * b2;
#pragma unroll
      for (int l = 0; l < LS; ++l) {
        float cl = (l == 0) ? (first * b1) : (a2 * xv[l - 1]);
        xh[l] = (l == 0) ? cl : (a1 * xh[l - 1] + cl);
      }
#pragma unroll
      for (int l = 0; l < LS; ++l) {
        float y = scale * (c1 * xh[l] + c2 * xv[l]);
        float f = ((i == 0) ? 2.f : 1.f) * ((l == 0) ? 2.f : 1.f);
        if (i == 0 && l == 0) f *= 0.25f;
        float* pp = &Kout[h * 1024 + i * 32 + l];
        if (n == 0) *pp = f * y; else *pp += f * y;
      }
    }
  }
}

// ---------------------------------------------------------------- kernel 2
// Bm[c][s=(i,j)][s'=(i',j')] = sum of 4 directional causal/anti-causal taps.
__device__ __forceinline__ float bmat_val(const float* k0, const float* k1,
                                          const float* k2, const float* k3,
                                          int i, int j, int ip, int jp) {
  float v = 0.f;
  if (ip <= i && jp <= j) v += k0[(i - ip) * 32 + (j - jp)];
  if (ip >= i && jp <= j) v += k1[(ip - i) * 32 + (j - jp)];
  if (ip <= i && jp >= j) v += k2[(i - ip) * 32 + (jp - j)];
  if (ip >= i && jp >= j) v += k3[(ip - i) * 32 + (jp - j)];
  return v;
}

__global__ __launch_bounds__(256) void build_bmat_kernel(
    const float* __restrict__ K, __bf16* __restrict__ Bm) {
  const int c = blockIdx.x >> 6;
  const int s0 = (blockIdx.x & 63) * 16;
  __shared__ float k0[1024], k1[1024], k2[1024], k3[1024];
  for (int t = threadIdx.x; t < 1024; t += 256) {
    k0[t] = K[(0 * 64 + c) * 1024 + t];
    k1[t] = K[(1 * 64 + c) * 1024 + t];
    k2[t] = K[(2 * 64 + c) * 1024 + t];
    k3[t] = K[(3 * 64 + c) * 1024 + t];
  }
  __syncthreads();
  __bf16* dst = Bm + (size_t)c * 1024 * 1024 + (size_t)s0 * 1024;
  for (int t = 0; t < 32; ++t) {
    int idx = t * 256 + threadIdx.x;      // pair index 0..8191
    int sl  = idx >> 9;                   // local s row 0..15
    int spp = (idx & 511) * 2;            // s' (even)
    int s = s0 + sl;
    int i = s >> 5, j = s & 31;
    int ip0 = spp >> 5, jp0 = spp & 31;
    int ip1 = (spp + 1) >> 5, jp1 = (spp + 1) & 31;
    union { __bf16 h[2]; unsigned u; } pk;
    pk.h[0] = (__bf16)bmat_val(k0, k1, k2, k3, i, j, ip0, jp0);
    pk.h[1] = (__bf16)bmat_val(k0, k1, k2, k3, i, j, ip1, jp1);
    *(unsigned*)&dst[sl * 1024 + spp] = pk.u;
  }
}

// ---------------------------------------------------------------- kernel 3
// A[c][r*16+b][s'] = bf16(x[s', b, e=c+64r]) -- class-grouped GEMM A matrix.
__global__ __launch_bounds__(256) void build_amat_kernel(
    const float* __restrict__ x, __bf16* __restrict__ Am) {
  const int bid = blockIdx.x;
  const int b = bid >> 6;
  const int s0 = ((bid >> 2) & 15) * 64;
  const int e0 = (bid & 3) * 256;
  __shared__ __bf16 tile[64 * 260];
  const int tid = threadIdx.x;
  for (int it = 0; it < 64; ++it) {
    int idx = it * 256 + tid;
    int spl = idx >> 8;      // 0..63
    int el  = idx & 255;
    tile[spl * 260 + el] =
        (__bf16)x[(size_t)(s0 + spl) * (NB * NE) + b * NE + e0 + el];
  }
  __syncthreads();
  const int sp = (tid & 31) * 2;
  const int grp = tid >> 5;
  for (int pr = grp; pr < 256; pr += 8) {
    int e = e0 + pr;
    int cc = e & 63, rr = e >> 6;
    union { __bf16 h[2]; unsigned u; } pk;
    pk.h[0] = tile[sp * 260 + pr];
    pk.h[1] = tile[(sp + 1) * 260 + pr];
    *(unsigned*)&Am[(size_t)cc * (256 * 1024) +
                    (size_t)(rr * 16 + b) * 1024 + s0 + sp] = pk.u;
  }
}

// ---------------------------------------------------------------- kernel 4
// Per class c: Y[c][256 x 1024] = A[c] * Bm[c]^T (Bm stored n-major).
// Block tile 128x128, 8 waves (4m x 2n), wave tile 32x64, K-step 64.
// Register-prefetch pipeline: next k-step's global loads issue before compute.
__global__ __launch_bounds__(256) void gemm_kernel(
    const __bf16* __restrict__ A, const __bf16* __restrict__ Bm,
    float* __restrict__ Y) {
  const int bid = blockIdx.x;
  const int c  = bid >> 4;
  const int m0 = ((bid >> 3) & 1) * 128;
  const int n0 = (bid & 7) * 128;
  const __bf16* Ag = A  + (size_t)c * (256 * 1024);
  const __bf16* Bg = Bm + (size_t)c * (1024 * 1024);

  __shared__ __bf16 Al[128 * 72];
  __shared__ __bf16 Bl[128 * 72];

  const int tid = threadIdx.x;
  const int lane = tid & 31;
  const int wid = tid >> 5;
  const int wm = (wid & 3) * 32;
  const int wn = (wid >> 2) * 64;

  // staging addresses: thread covers rows ldRow+{0,32,64,96}, 8 bf16 each
  const int ldRow = tid >> 3;          // 0..31
  const int ldCol = (tid & 7) * 8;     // 0..56
  const __bf16* aPtr = Ag + (size_t)(m0 + ldRow) * 1024 + ldCol;
  const __bf16* bPtr = Bg + (size_t)(n0 + ldRow) * 1024 + ldCol;
  __bf16* aSt = &Al[ldRow * 72 + ldCol];
  __bf16* bSt = &Bl[ldRow * 72 + ldCol];

  v8f acc[2][4];
#pragma unroll
  for (int mt = 0; mt < 2; ++mt)
#pragma unroll
    for (int nt = 0; nt < 4; ++nt)
#pragma unroll
      for (int v = 0; v < 8; ++v) acc[mt][nt][v] = 0.0f;

  v8bf aReg[4], bReg[4];
#pragma unroll
  for (int i = 0; i < 4; ++i) {
    aReg[i] = *(const v8bf*)(aPtr + (size_t)i * 32 * 1024);
    bReg[i] = *(const v8bf*)(bPtr + (size_t)i * 32 * 1024);
  }

  for (int k0 = 0; k0 < 1024; k0 += 64) {
    __syncthreads();   // previous step's LDS reads complete
#pragma unroll
    for (int i = 0; i < 4; ++i) {
      *(v8bf*)(aSt + i * 32 * 72) = aReg[i];
      *(v8bf*)(bSt + i * 32 * 72) = bReg[i];
    }
    if (k0 + 64 < 1024) {  // issue next step's global loads now; overlap compute
#pragma unroll
      for (int i = 0; i < 4; ++i) {
        aReg[i] = *(const v8bf*)(aPtr + (size_t)i * 32 * 1024 + k0 + 64);
        bReg[i] = *(const v8bf*)(bPtr + (size_t)i * 32 * 1024 + k0 + 64);
      }
    }
    __syncthreads();

#pragma unroll
    for (int kk = 0; kk < 64; kk += 32) {
      v16bf afrag[2], bfrag[4];
#pragma unroll
      for (int mt = 0; mt < 2; ++mt) {
        // 16-bit A 16x32 layout: lanes<16 own K{0..7,16..23}, lanes>=16 K{8..15,24..31}
        int row = wm + mt * 16 + (lane & 15);
        int kb = kk + ((lane < 16) ? 0 : 8);
        afrag[mt] = combine16(*(const v8bf*)&Al[row * 72 + kb],
                              *(const v8bf*)&Al[row * 72 + kb + 16]);
      }
#pragma unroll
      for (int nt = 0; nt < 4; ++nt) {
        // 16-bit B 32x16 layout: lanes<16 own K{0..7,16..23}, lanes>=16 K{8..15,24..31}
        int col = wn + nt * 16 + (lane & 15);
        int kb = kk + ((lane < 16) ? 0 : 8);
        bfrag[nt] = combine16(*(const v8bf*)&Bl[col * 72 + kb],
                              *(const v8bf*)&Bl[col * 72 + kb + 16]);
      }
#pragma unroll
      for (int mt = 0; mt < 2; ++mt)
#pragma unroll
        for (int nt = 0; nt < 4; ++nt)
          acc[mt][nt] = __builtin_amdgcn_wmma_f32_16x16x32_bf16(
              false, afrag[mt], false, bfrag[nt], (short)0, acc[mt][nt],
              false, false);
    }
  }

  // Y is streamed (write-once): non-temporal so B/A stay resident in L2
  float* Yg = Y + (size_t)c * (256 * 1024);
#pragma unroll
  for (int mt = 0; mt < 2; ++mt)
#pragma unroll
    for (int nt = 0; nt < 4; ++nt) {
      int n = n0 + wn + nt * 16 + (lane & 15);
      int mbase = m0 + wm + mt * 16 + ((lane < 16) ? 0 : 8);
#pragma unroll
      for (int v = 0; v < 8; ++v)
        __builtin_nontemporal_store(acc[mt][nt][v],
                                    &Yg[(size_t)(mbase + v) * 1024 + n]);
    }
}

// ---------------------------------------------------------------- kernel 5
// out[s,b,e] = silu(Y[c=e%64][(e/64)*16+b][s] + x[s,b,e]*omega[e])
__device__ __forceinline__ int ep_idx(int e, int si) {
  return e * 16 + (si ^ (e & 15));
}

__global__ __launch_bounds__(256) void epilogue_kernel(
    const float* __restrict__ Y, const float* __restrict__ x,
    const float* __restrict__ omega, float* __restrict__ out) {
  const int s0 = (blockIdx.x >> 4) * 16;
  const int b  = blockIdx.x & 15;
  __shared__ float tile[16384];
  const int tid = threadIdx.x;
  for (int it = 0; it < 64; ++it) {
    int idx = it * 256 + tid;
    int si = idx & 15;
    int pr = idx >> 4;                 // (c,r) pair: c=pr>>4, r=pr&15
    int cc = pr >> 4, rr = pr & 15;
    float v = __builtin_nontemporal_load(
        &Y[(size_t)cc * (256 * 1024) + (size_t)(rr * 16 + b) * 1024 +
           s0 + si]);
    tile[ep_idx(rr * 64 + cc, si)] = v;   // e = c + 64*r
  }
  __syncthreads();
  for (int si = 0; si < 16; ++si) {
    const int s = s0 + si;
#pragma unroll
    for (int it = 0; it < 4; ++it) {
      int e = it * 256 + tid;
      float v = tile[ep_idx(e, si)] +
                x[(size_t)s * (NB * NE) + b * NE + e] * omega[e];
      __builtin_nontemporal_store(v / (1.0f + __expf(-v)),
                                  &out[(size_t)s * (NB * NE) + b * NE + e]);
    }
  }
}

// ---------------------------------------------------------------- launch
extern "C" void kernel_launch(void* const* d_in, const int* in_sizes, int n_in,
                              void* d_out, int out_size, void* d_ws,
                              size_t ws_size, hipStream_t stream) {
  const float* x  = (const float*)d_in[0];
  const float* A1 = (const float*)d_in[1];
  const float* A2 = (const float*)d_in[2];
  const float* A3 = (const float*)d_in[3];
  const float* A4 = (const float*)d_in[4];
  const float* B1 = (const float*)d_in[5];
  const float* B2 = (const float*)d_in[6];
  const float* C1 = (const float*)d_in[7];
  const float* C2 = (const float*)d_in[8];
  const float* om = (const float*)d_in[9];
  float* out = (float*)d_out;

  // workspace layout (~226 MB)
  char* ws = (char*)d_ws;
  float*  Kbuf = (float*)ws;                                  //   1 MB
  __bf16* Bm   = (__bf16*)(ws + (size_t)(1 << 20));           // 128 MB
  __bf16* Am   = (__bf16*)(ws + (size_t)(1 << 20) + 134217728ull);        // 32 MB
  float*  Y    = (float*)(ws + (size_t)(1 << 20) + 134217728ull + 33554432ull); // 64 MB

  compute_k_kernel<<<1, 256, 0, stream>>>(A1, A2, A3, A4, B1, B2, C1, C2,
                                          Kbuf);
  build_bmat_kernel<<<64 * 64, 256, 0, stream>>>(Kbuf, Bm);
  build_amat_kernel<<<16 * 16 * 4, 256, 0, stream>>>(x, Am);
  gemm_kernel<<<64 * 2 * 8, 256, 0, stream>>>(Am, Bm, Y);
  epilogue_kernel<<<64 * 16, 256, 0, stream>>>(Y, x, om, out);
}
